// TalkingHeadsAttention_21852793602258
// MI455X (gfx1250) — matrix operand order repaired and checked by reading, not compiled, not run on hardware
//
#include <hip/hip_runtime.h>

// ---------------- Talking-heads attention, MI455X / gfx1250, wave32 WMMA ----
// Pipeline (all bf16 WMMA, f32 accumulate):
//   K0a: x -> bf16            (A-side data converted once)
//   K0b: w_qkv, w_out -> bf16 packed in WMMA B-fragment order (contig loads)
//   K1 : qkv = x @ w_qkv  -> Q[b,h,n,d], K[b,h,n,d], VT[b,h,d,n]  (bf16)
//   K2 : per (b, 16-row tile): flash pass 1 -> row max m, row sum l per head
//        (raw QK^T tiles per head via WMMA, pre-mix across heads through LDS)
//   K3 : flash pass 2 -> recompute scores, P_h = exp(s-m)/l, post-mix with
//        w_post straight into WMMA A-fragments, accumulate P @ V -> OH bf16
//   K4 : out = OH @ w_out + b_out  (f32 result)

typedef __bf16 v16bf __attribute__((ext_vector_type(16)));
typedef float  v8f   __attribute__((ext_vector_type(8)));

#define NDIM   512
#define NHEADS 8
#define DHEAD  64
#define NINNER 512
#define NCTX   2048
#define NBATCH 2
#define QKSCALE 0.125f   // 1/sqrt(64)
#define NKB    (NDIM / 32)   // 16 K-blocks for both GEMMs (K = 512)

static __device__ __forceinline__ v8f wmma_bf16(v16bf a, v16bf b, v8f c) {
  return __builtin_amdgcn_wmma_f32_16x16x32_bf16(false, a, false, b,
                                                 (short)0, c, false, false);
}

// A fragment (16x32, bf16): lane = row (M = lane%16); element e -> K index:
//   K = 16*(lane/16)+e (contiguous, matches B fragments)
template <typename T>
static __device__ __forceinline__ v16bf load_a(const T* __restrict__ src,
                                               int ld, int row0, int k0,
                                               int half, int lr) {
  v16bf a;
  const T* p = src + (size_t)(row0 + lr) * ld + k0;
#pragma unroll
  for (int e = 0; e < 8; ++e) a[e] = (__bf16)p[16 * half + e];
#pragma unroll
  for (int e = 0; e < 8; ++e) a[e + 8] = (__bf16)p[16 * half + 8 + e];
  return a;
}

// B fragment from 16 contiguous bf16 (packed weights / K^T / V^T layouts)
static __device__ __forceinline__ v16bf load_b_contig(
    const __bf16* __restrict__ p) {
  v16bf b;
#pragma unroll
  for (int e = 0; e < 16; ++e) b[e] = p[e];
  return b;
}

// ---------------- K0a: x f32 -> bf16 ------------------------------------
__global__ __launch_bounds__(256) void k_cvt_x(const float* __restrict__ x,
                                               __bf16* __restrict__ xb,
                                               int n) {
  int i = (blockIdx.x * 256 + threadIdx.x) * 4;
  if (i + 3 < n) {
#pragma unroll
    for (int j = 0; j < 4; ++j) xb[i + j] = (__bf16)x[i + j];
  }
}

// ---------------- K0b: pack weights into B-fragment order ----------------
// Layout: wp[((ctile*NKB + kb)*32 + lane)*16 + e]
//   = (bf16) w[(kb*32 + 16*(lane/16) + e) * ncols + ctile*16 + (lane&15)]
__global__ __launch_bounds__(256) void k_packw(const float* __restrict__ w,
                                               __bf16* __restrict__ wp,
                                               int ncols) {
  const int tid = blockIdx.x * 256 + threadIdx.x;   // one (ctile,kb,lane)
  const int lane = tid & 31;
  const int kb = (tid >> 5) & (NKB - 1);
  const int ctile = tid / (32 * NKB);
  if (ctile >= ncols / 16) return;
  const int half = lane >> 4, lr = lane & 15;
  __bf16* dst = wp + (size_t)tid * 16;
#pragma unroll
  for (int e = 0; e < 16; ++e)
    dst[e] =
        (__bf16)w[(size_t)(kb * 32 + 16 * half + e) * ncols + ctile * 16 + lr];
}

// ---------------- K1: QKV projection -------------------------------------
__global__ __launch_bounds__(256) void k_qkv(
    const __bf16* __restrict__ xb, const __bf16* __restrict__ wqkvp,
    __bf16* __restrict__ Q, __bf16* __restrict__ K, __bf16* __restrict__ VT) {
  const int lane = threadIdx.x & 31, wave = threadIdx.x >> 5;
  const int half = lane >> 4, lr = lane & 15;
  const int tile = blockIdx.x * 8 + wave;   // 256 * 96 = 24576 tiles
  const int itile = tile / 96, ctile = tile % 96;
  const int row0 = itile * 16, c0 = ctile * 16;

  const __bf16* bsrc = wqkvp + (((size_t)ctile * NKB) * 32 + lane) * 16;
  v8f acc = {};
#pragma unroll 4
  for (int kb = 0; kb < NKB; ++kb) {
    v16bf a = load_a(xb, NDIM, row0, kb * 32, half, lr);
    v16bf b = load_b_contig(bsrc + (size_t)kb * 32 * 16);
    acc = wmma_bf16(a, b, acc);
  }
  const int which = c0 / NINNER;           // 0=Q 1=K 2=V (uniform per wave)
  const int rem = c0 % NINNER;
  const int h = rem / DHEAD;
  const int dd = (rem % DHEAD) + lr;
  const int batch = row0 >> 11;
  const int nbase = row0 & (NCTX - 1);
  if (which < 2) {
    __bf16* dst = which == 0 ? Q : K;
#pragma unroll
    for (int r = 0; r < 8; ++r) {
      int n = nbase + r + 8 * half;
      dst[(((size_t)(batch * NHEADS + h)) * NCTX + n) * DHEAD + dd] =
          (__bf16)acc[r];
    }
  } else {
#pragma unroll
    for (int r = 0; r < 8; ++r) {
      int n = nbase + r + 8 * half;
      VT[(((size_t)(batch * NHEADS + h)) * DHEAD + dd) * NCTX + n] =
          (__bf16)acc[r];
    }
  }
}

// ---------------- K2: softmax row stats (flash pass 1) -------------------
__global__ __launch_bounds__(256) void k_stats(
    const __bf16* __restrict__ Q, const __bf16* __restrict__ Km,
    const float* __restrict__ wpre, float* __restrict__ Ms,
    float* __restrict__ Ls) {
  __shared__ float rawS[NHEADS][16][32];
  const int lane = threadIdx.x & 31, h = threadIdx.x >> 5;  // wave == head
  const int half = lane >> 4, lr = lane & 15;
  const int batch = blockIdx.x >> 7;
  const int i0 = (blockIdx.x & 127) * 16;

  float wp[8];
#pragma unroll
  for (int j = 0; j < 8; ++j) wp[j] = wpre[h * 8 + j];

  const __bf16* Qh = Q + ((size_t)(batch * NHEADS + h)) * NCTX * DHEAD;
  const __bf16* Kh = Km + ((size_t)(batch * NHEADS + h)) * NCTX * DHEAD;
  const v16bf aq0 = load_a(Qh, DHEAD, i0, 0, half, lr);
  const v16bf aq1 = load_a(Qh, DHEAD, i0, 32, half, lr);

  float m[8], l[8];
#pragma unroll
  for (int r = 0; r < 8; ++r) { m[r] = -3.0e38f; l[r] = 0.f; }

  for (int jt = 0; jt < NCTX / 32; ++jt) {
    const int j0 = jt * 32;
    v8f s0 = {}, s1 = {};
    {
      const __bf16* kb0 = Kh + (size_t)(j0 + lr) * DHEAD + 16 * half;
      s0 = wmma_bf16(aq0, load_b_contig(kb0), s0);
      s0 = wmma_bf16(aq1, load_b_contig(kb0 + 32), s0);
      const __bf16* kb1 = Kh + (size_t)(j0 + 16 + lr) * DHEAD + 16 * half;
      s1 = wmma_bf16(aq0, load_b_contig(kb1), s1);
      s1 = wmma_bf16(aq1, load_b_contig(kb1 + 32), s1);
    }
#pragma unroll
    for (int r = 0; r < 8; ++r) {
      rawS[h][r + 8 * half][lr] = s0[r];
      rawS[h][r + 8 * half][16 + lr] = s1[r];
    }
    __syncthreads();
#pragma unroll
    for (int r = 0; r < 8; ++r) {
      const int Mr = r + 8 * half;
      float sm0 = 0.f, sm1 = 0.f;
#pragma unroll
      for (int hh = 0; hh < 8; ++hh) {
        sm0 += wp[hh] * rawS[hh][Mr][lr];
        sm1 += wp[hh] * rawS[hh][Mr][16 + lr];
      }
      sm0 *= QKSCALE; sm1 *= QKSCALE;
      float x = fmaxf(sm0, sm1);
#pragma unroll
      for (int d = 1; d < 16; d <<= 1) x = fmaxf(x, __shfl_xor(x, d, 32));
      const float mn = fmaxf(m[r], x);
      float p = __expf(sm0 - mn) + __expf(sm1 - mn);
#pragma unroll
      for (int d = 1; d < 16; d <<= 1) p += __shfl_xor(p, d, 32);
      l[r] = l[r] * __expf(m[r] - mn) + p;
      m[r] = mn;
    }
    __syncthreads();
  }
  if (lr == 0) {
#pragma unroll
    for (int r = 0; r < 8; ++r) {
      size_t idx = ((size_t)(batch * NHEADS + h)) * NCTX + i0 + r + 8 * half;
      Ms[idx] = m[r];
      Ls[idx] = l[r];
    }
  }
}

// ---------------- K3: flash pass 2, post-mix, P @ V ----------------------
__global__ __launch_bounds__(256) void k_out(
    const __bf16* __restrict__ Q, const __bf16* __restrict__ Km,
    const __bf16* __restrict__ VT, const float* __restrict__ wpre,
    const float* __restrict__ wpost, const float* __restrict__ Ms,
    const float* __restrict__ Ls, __bf16* __restrict__ OH) {
  __shared__ float rawS[NHEADS][16][32];
  __shared__ float Pl[NHEADS][16][32];
  const int lane = threadIdx.x & 31, g = threadIdx.x >> 5;  // wave == head
  const int half = lane >> 4, lr = lane & 15;
  const int batch = blockIdx.x >> 7;
  const int i0 = (blockIdx.x & 127) * 16;

  float wp[8], wq[8];
#pragma unroll
  for (int j = 0; j < 8; ++j) { wp[j] = wpre[g * 8 + j]; wq[j] = wpost[g * 8 + j]; }

  float mg[8], il[8];
#pragma unroll
  for (int r = 0; r < 8; ++r) {
    size_t idx = ((size_t)(batch * NHEADS + g)) * NCTX + i0 + r + 8 * half;
    mg[r] = Ms[idx];
    il[r] = 1.0f / Ls[idx];
  }

  const __bf16* Qh = Q + ((size_t)(batch * NHEADS + g)) * NCTX * DHEAD;
  const __bf16* Kh = Km + ((size_t)(batch * NHEADS + g)) * NCTX * DHEAD;
  const __bf16* Vh = VT + ((size_t)(batch * NHEADS + g)) * DHEAD * NCTX;
  const v16bf aq0 = load_a(Qh, DHEAD, i0, 0, half, lr);
  const v16bf aq1 = load_a(Qh, DHEAD, i0, 32, half, lr);

  v8f accd[4] = {};
  for (int jt = 0; jt < NCTX / 32; ++jt) {
    const int j0 = jt * 32;
    v8f s0 = {}, s1 = {};
    {
      const __bf16* kb0 = Kh + (size_t)(j0 + lr) * DHEAD + 16 * half;
      s0 = wmma_bf16(aq0, load_b_contig(kb0), s0);
      s0 = wmma_bf16(aq1, load_b_contig(kb0 + 32), s0);
      const __bf16* kb1 = Kh + (size_t)(j0 + 16 + lr) * DHEAD + 16 * half;
      s1 = wmma_bf16(aq0, load_b_contig(kb1), s1);
      s1 = wmma_bf16(aq1, load_b_contig(kb1 + 32), s1);
    }
#pragma unroll
    for (int r = 0; r < 8; ++r) {
      rawS[g][r + 8 * half][lr] = s0[r];
      rawS[g][r + 8 * half][16 + lr] = s1[r];
    }
    __syncthreads();
    // pre-mix + per-head normalized P for this wave's head
#pragma unroll
    for (int r = 0; r < 8; ++r) {
      const int Mr = r + 8 * half;
      float sm0 = 0.f, sm1 = 0.f;
#pragma unroll
      for (int hh = 0; hh < 8; ++hh) {
        sm0 += wp[hh] * rawS[hh][Mr][lr];
        sm1 += wp[hh] * rawS[hh][Mr][16 + lr];
      }
      Pl[g][Mr][lr] = __expf(sm0 * QKSCALE - mg[r]) * il[r];
      Pl[g][Mr][16 + lr] = __expf(sm1 * QKSCALE - mg[r]) * il[r];
    }
    __syncthreads();
    // post-mix straight into WMMA A-fragment layout
    v16bf pa;
#pragma unroll
    for (int e = 0; e < 16; ++e) {
      const int k = 16 * half + e;
      float v = 0.f;
#pragma unroll
      for (int hh = 0; hh < 8; ++hh) v += wq[hh] * Pl[hh][lr][k];
      pa[e] = (__bf16)v;
    }
    const __bf16* vbase = Vh + (size_t)lr * NCTX + j0 + 16 * half;
#pragma unroll
    for (int t2 = 0; t2 < 4; ++t2) {
      v16bf bv = load_b_contig(vbase + (size_t)(16 * t2) * NCTX);
      accd[t2] = wmma_bf16(pa, bv, accd[t2]);
    }
  }
#pragma unroll
  for (int t2 = 0; t2 < 4; ++t2)
#pragma unroll
    for (int r = 0; r < 8; ++r) {
      const int n = i0 + r + 8 * half;
      OH[((size_t)batch * NCTX + n) * NINNER + g * DHEAD + 16 * t2 + lr] =
          (__bf16)accd[t2][r];
    }
}

// ---------------- K4: output projection + bias ---------------------------
__global__ __launch_bounds__(256) void k_proj(
    const __bf16* __restrict__ OH, const __bf16* __restrict__ woutp,
    const float* __restrict__ bout, float* __restrict__ out) {
  const int lane = threadIdx.x & 31, wave = threadIdx.x >> 5;
  const int half = lane >> 4, lr = lane & 15;
  const int tile = blockIdx.x * 8 + wave;  // 256*32 = 8192 tiles
  const int row0 = (tile / 32) * 16, ctile = tile % 32;
  const int c0 = ctile * 16;

  const __bf16* bsrc = woutp + (((size_t)ctile * NKB) * 32 + lane) * 16;
  v8f acc = {};
#pragma unroll 4
  for (int kb = 0; kb < NKB; ++kb) {
    v16bf a = load_a(OH, NINNER, row0, kb * 32, half, lr);
    v16bf b = load_b_contig(bsrc + (size_t)kb * 32 * 16);
    acc = wmma_bf16(a, b, acc);
  }
  const float bb = bout[c0 + lr];
#pragma unroll
  for (int r = 0; r < 8; ++r)
    out[(size_t)(row0 + r + 8 * half) * NDIM + c0 + lr] = acc[r] + bb;
}

extern "C" void kernel_launch(void* const* d_in, const int* in_sizes, int n_in,
                              void* d_out, int out_size, void* d_ws,
                              size_t ws_size, hipStream_t stream) {
  const float* x     = (const float*)d_in[0];
  const float* wqkv  = (const float*)d_in[1];
  const float* wpre  = (const float*)d_in[2];
  const float* wpost = (const float*)d_in[3];
  const float* wout  = (const float*)d_in[4];
  const float* bout  = (const float*)d_in[5];
  float* out = (float*)d_out;

  char* ws = (char*)d_ws;
  const size_t qkvElems = (size_t)NBATCH * NHEADS * NCTX * DHEAD;  // 2 Mi
  const size_t ohElems  = (size_t)NBATCH * NCTX * NINNER;          // 2 Mi
  const size_t xElems   = (size_t)NBATCH * NCTX * NDIM;            // 2 Mi
  size_t off = 0;
  __bf16* Q   = (__bf16*)(ws + off); off += qkvElems * 2;
  __bf16* K   = (__bf16*)(ws + off); off += qkvElems * 2;
  __bf16* VT  = (__bf16*)(ws + off); off += qkvElems * 2;
  __bf16* OH  = (__bf16*)(ws + off); off += ohElems * 2;
  __bf16* Xb  = (__bf16*)(ws + off); off += xElems * 2;
  __bf16* Wqp = (__bf16*)(ws + off); off += (size_t)NDIM * 3 * NINNER * 2;
  __bf16* Wop = (__bf16*)(ws + off); off += (size_t)NINNER * NDIM * 2;
  float* Ms   = (float*)(ws + off);  off += (size_t)NBATCH * NHEADS * NCTX * 4;
  float* Ls   = (float*)(ws + off);  off += (size_t)NBATCH * NHEADS * NCTX * 4;

  k_cvt_x<<<(int)(xElems / 4 / 256), 256, 0, stream>>>(x, Xb, (int)xElems);
  k_packw<<<(96 * NKB * 32) / 256, 256, 0, stream>>>(wqkv, Wqp, 3 * NINNER);
  k_packw<<<(32 * NKB * 32) / 256, 256, 0, stream>>>(wout, Wop, NDIM);

  k_qkv <<<3072, 256, 0, stream>>>(Xb, Wqp, Q, K, VT);
  k_stats<<<NBATCH * 128, 256, 0, stream>>>(Q, K, wpre, Ms, Ls);
  k_out <<<NBATCH * 128, 256, 0, stream>>>(Q, K, VT, wpre, wpost, Ms, Ls, OH);
  k_proj<<<1024, 256, 0, stream>>>(OH, Wop, bout, out);
}
